// QuantLinearNet_8211977470370
// MI455X (gfx1250) — 8-bit matrix operand order repaired and checked by reading, compiled, not run on hardware
//
#include <hip/hip_runtime.h>
#include <hip/hip_bf16.h>


typedef __attribute__((ext_vector_type(8))) int v8i;
typedef int qln_v4i __attribute__((vector_size(16)));

#define TILE_M 128
#define TILE_N 128
#define TILE_K 64

#if __has_builtin(__builtin_amdgcn_global_load_async_to_lds_b128)
#define QLN_ASYNC_LDS 1
#else
#define QLN_ASYNC_LDS 0
#endif

#define AS1 __attribute__((address_space(1)))
#define AS3 __attribute__((address_space(3)))


__global__ void qln_init_kernel(unsigned int* meta) {
    if (threadIdx.x == 0) { meta[0] = 0u; meta[1] = 0u; meta[2] = 0u; }
}

__global__ void qln_absmax_kernel(const float* __restrict__ in, size_t n,
                                  unsigned int* __restrict__ out_bits) {
    __shared__ float red[256];
    float m = 0.0f;
    for (size_t i = (size_t)blockIdx.x * blockDim.x + threadIdx.x; i < n;
         i += (size_t)gridDim.x * blockDim.x) {
        m = fmaxf(m, fabsf(in[i]));
    }
    red[threadIdx.x] = m;
    __syncthreads();
    for (int s = 128; s > 0; s >>= 1) {
        if ((int)threadIdx.x < s) red[threadIdx.x] = fmaxf(red[threadIdx.x], red[threadIdx.x + s]);
        __syncthreads();
    }
    if (threadIdx.x == 0) atomicMax(out_bits, __float_as_uint(red[0]));
}

__global__ void qln_finalize_scales(const unsigned int* __restrict__ meta,
                                    float* __restrict__ scl) {
    if (threadIdx.x == 0) {
        float s_in = __uint_as_float(meta[0]) / 127.0f;
        float s_w  = __uint_as_float(meta[1]) / 7.0f;
        scl[0] = s_in;
        scl[1] = s_w;
        scl[2] = s_in * s_w;
    }
}

__global__ void qln_quant_kernel(const float* __restrict__ in,
                                 signed char* __restrict__ out,
                                 const float* __restrict__ scl, int sidx,
                                 float qmax, size_t n4) {
    float s = scl[sidx];
    float inv = 1.0f / s;
    int* __restrict__ out32 = (int*)out;
    const float4* __restrict__ in4 = (const float4*)in;
    for (size_t i = (size_t)blockIdx.x * blockDim.x + threadIdx.x; i < n4;
         i += (size_t)gridDim.x * blockDim.x) {
        float4 v = in4[i];
        int q0 = (int)fminf(fmaxf(rintf(v.x * inv), -qmax), qmax);
        int q1 = (int)fminf(fmaxf(rintf(v.y * inv), -qmax), qmax);
        int q2 = (int)fminf(fmaxf(rintf(v.z * inv), -qmax), qmax);
        int q3 = (int)fminf(fmaxf(rintf(v.w * inv), -qmax), qmax);
        out32[i] = (q0 & 255) | ((q1 & 255) << 8) | ((q2 & 255) << 16) | ((q3 & 255) << 24);
    }
}

__global__ void qln_outquant_kernel(float* __restrict__ y,
                                    const unsigned int* __restrict__ amax_bits,
                                    size_t n4) {
    float s = __uint_as_float(*amax_bits) / 127.0f;
    float inv = 1.0f / s;
    float4* __restrict__ y4 = (float4*)y;
    for (size_t i = (size_t)blockIdx.x * blockDim.x + threadIdx.x; i < n4;
         i += (size_t)gridDim.x * blockDim.x) {
        float4 v = y4[i];
        v.x = fminf(fmaxf(rintf(v.x * inv), -127.0f), 127.0f) * s;
        v.y = fminf(fmaxf(rintf(v.y * inv), -127.0f), 127.0f) * s;
        v.z = fminf(fmaxf(rintf(v.z * inv), -127.0f), 127.0f) * s;
        v.w = fminf(fmaxf(rintf(v.w * inv), -127.0f), 127.0f) * s;
        y4[i] = v;
    }
}

__global__ __launch_bounds__(256)
void qln_gemm_kernel(const signed char* __restrict__ xq,
                     const signed char* __restrict__ wq,
                     const float* __restrict__ bias,
                     const float* __restrict__ scl,
                     float* __restrict__ y,
                     unsigned int* __restrict__ amax_y_bits,
                     int M, int N, int K) {
    __shared__ __align__(16) signed char As[2][TILE_M * TILE_K];
    __shared__ __align__(16) signed char Bs[2][TILE_N * TILE_K];
    __shared__ float red[256];

    const int tid  = threadIdx.x;
    const int lane = tid & 31;
    const int wave = tid >> 5;
    const int wx   = wave & 3;
    const int wy   = wave >> 2;
    const int l15  = lane & 15;
    const int half = lane >> 4;

    const int block_m = blockIdx.y * TILE_M;
    const int block_n = blockIdx.x * TILE_N;

    const int row0 = tid >> 2,          off0 = (tid & 3) * 16;
    const int row1 = (tid + 256) >> 2,  off1 = ((tid + 256) & 3) * 16;

    auto stage = [&](int buf, int k0) {
#if QLN_ASYNC_LDS
        __builtin_amdgcn_global_load_async_to_lds_b128(
            (AS1 qln_v4i*)(&xq[(size_t)(block_m + row0) * K + k0 + off0]),
            (AS3 qln_v4i*)(&As[buf][row0 * TILE_K + off0]), 0, 0);
        __builtin_amdgcn_global_load_async_to_lds_b128(
            (AS1 qln_v4i*)(&xq[(size_t)(block_m + row1) * K + k0 + off1]),
            (AS3 qln_v4i*)(&As[buf][row1 * TILE_K + off1]), 0, 0);
        __builtin_amdgcn_global_load_async_to_lds_b128(
            (AS1 qln_v4i*)(&wq[(size_t)(block_n + row0) * K + k0 + off0]),
            (AS3 qln_v4i*)(&Bs[buf][row0 * TILE_K + off0]), 0, 0);
        __builtin_amdgcn_global_load_async_to_lds_b128(
            (AS1 qln_v4i*)(&wq[(size_t)(block_n + row1) * K + k0 + off1]),
            (AS3 qln_v4i*)(&Bs[buf][row1 * TILE_K + off1]), 0, 0);
#else
        uint4 a0 = *(const uint4*)(&xq[(size_t)(block_m + row0) * K + k0 + off0]);
        uint4 a1 = *(const uint4*)(&xq[(size_t)(block_m + row1) * K + k0 + off1]);
        uint4 b0 = *(const uint4*)(&wq[(size_t)(block_n + row0) * K + k0 + off0]);
        uint4 b1 = *(const uint4*)(&wq[(size_t)(block_n + row1) * K + k0 + off1]);
        *(uint4*)(&As[buf][row0 * TILE_K + off0]) = a0;
        *(uint4*)(&As[buf][row1 * TILE_K + off1]) = a1;
        *(uint4*)(&Bs[buf][row0 * TILE_K + off0]) = b0;
        *(uint4*)(&Bs[buf][row1 * TILE_K + off1]) = b1;
#endif
    };

    const v8i vzero = {0, 0, 0, 0, 0, 0, 0, 0};
    v8i acc[4][2];
#pragma unroll
    for (int i = 0; i < 4; ++i)
#pragma unroll
        for (int j = 0; j < 2; ++j) acc[i][j] = vzero;

    stage(0, 0);
#if QLN_ASYNC_LDS
    asm volatile("s_wait_asynccnt 0x0" ::: "memory");
#endif
    __syncthreads();

    int buf = 0;
    for (int k0 = 0; k0 < K; k0 += TILE_K) {
        if (k0 + TILE_K < K) {
            stage(buf ^ 1, k0 + TILE_K);
            if (k0 + 2 * TILE_K < K) {
                __builtin_prefetch(&xq[(size_t)(block_m + (tid >> 1)) * K + k0 + 2 * TILE_K], 0, 3);
                __builtin_prefetch(&wq[(size_t)(block_n + (tid >> 1)) * K + k0 + 2 * TILE_K], 0, 3);
            }
        }

        v8i afrag[4], bfrag[2];
#pragma unroll
        for (int i = 0; i < 4; ++i) {
            int m = wy * 64 + i * 16 + l15;
            const int* ap = (const int*)(&As[buf][m * TILE_K]);
#pragma unroll
            for (int v = 0; v < 8; ++v) {
                int kb = ((v >> 1) << 4) + ((v & 1) << 2) + half * 8;
                afrag[i][v] = ap[kb >> 2];
            }
        }
#pragma unroll
        for (int j = 0; j < 2; ++j) {
            int n = wx * 32 + j * 16 + l15;
            const int* bp = (const int*)(&Bs[buf][n * TILE_K]);
#pragma unroll
            for (int v = 0; v < 8; ++v) {
                int kb = ((v >> 1) << 4) + ((v & 1) << 2) + half * 8;
                bfrag[j][v] = bp[kb >> 2];
            }
        }

#pragma unroll
        for (int i = 0; i < 4; ++i)
#pragma unroll
            for (int j = 0; j < 2; ++j)
                acc[i][j] = __builtin_amdgcn_wmma_i32_16x16x64_iu8(
                    true, afrag[i], true, bfrag[j], acc[i][j], false, false);

#if QLN_ASYNC_LDS
        asm volatile("s_wait_asynccnt 0x0" ::: "memory");
#endif
        __syncthreads();
        buf ^= 1;
    }

    const float s_b = scl[2];
    float lmax = 0.0f;
#pragma unroll
    for (int j = 0; j < 2; ++j) {
        int n = block_n + wx * 32 + j * 16 + l15;
        float bq = rintf(bias[n] / s_b);
        bq = fminf(fmaxf(bq, -2147483647.0f), 2147483647.0f) * s_b;
#pragma unroll
        for (int i = 0; i < 4; ++i) {
#pragma unroll
            for (int v = 0; v < 8; ++v) {
                int m = block_m + wy * 64 + i * 16 + v + half * 8;
                float val = (float)acc[i][j][v] * s_b + bq;
                y[(size_t)m * N + n] = val;
                lmax = fmaxf(lmax, fabsf(val));
            }
        }
    }
    red[tid] = lmax;
    __syncthreads();
    for (int s = 128; s > 0; s >>= 1) {
        if (tid < s) red[tid] = fmaxf(red[tid], red[tid + s]);
        __syncthreads();
    }
    if (tid == 0) atomicMax(amax_y_bits, __float_as_uint(red[0]));
}


extern "C" void kernel_launch(void* const* d_in, const int* in_sizes, int n_in,
                              void* d_out, int out_size, void* d_ws, size_t ws_size,
                              hipStream_t stream) {
    const float* x = (const float*)d_in[0];
    const float* W = (const float*)d_in[1];
    const float* b = (const float*)d_in[2];
    float* y = (float*)d_out;

    const int N = in_sizes[2];
    const int K = in_sizes[1] / N;
    const int M = (int)((long long)in_sizes[0] / K);

    unsigned int* meta = (unsigned int*)d_ws;
    float* scl = (float*)((char*)d_ws + 16);
    signed char* xq = (signed char*)d_ws + 256;
    signed char* wq = xq + (size_t)M * K;

    qln_init_kernel<<<1, 64, 0, stream>>>(meta);
    qln_absmax_kernel<<<2048, 256, 0, stream>>>(x, (size_t)M * K, meta + 0);
    qln_absmax_kernel<<<1024, 256, 0, stream>>>(W, (size_t)N * K, meta + 1);
    qln_finalize_scales<<<1, 64, 0, stream>>>(meta, scl);
    qln_quant_kernel<<<4096, 256, 0, stream>>>(x, xq, scl, 0, 127.0f, (size_t)M * K / 4);
    qln_quant_kernel<<<2048, 256, 0, stream>>>(W, wq, scl, 1, 7.0f, (size_t)N * K / 4);

    dim3 grid(N / TILE_N, M / TILE_M);
    qln_gemm_kernel<<<grid, 256, 0, stream>>>(xq, wq, b, scl, y, meta + 2, M, N, K);

    qln_outquant_kernel<<<4096, 256, 0, stream>>>(y, meta + 2, (size_t)M * N / 4);
}
